// U_Mamba_18245021073490
// MI455X (gfx1250) — matrix operand order repaired and checked by reading, not compiled, not run on hardware
//
#include <hip/hip_runtime.h>
#include <cstddef>

// ---------------------------------------------------------------------------
// Types for CDNA5 WMMA (gfx1250, wave32)
// ---------------------------------------------------------------------------
typedef __attribute__((ext_vector_type(16))) __bf16 v16bf;
typedef __attribute__((ext_vector_type(8)))  __bf16 v8bf;
typedef __attribute__((ext_vector_type(8)))  float  v8f;

#define BATCH 4
#define CH    64

// ===========================================================================
// GEMM:  C[256][N] = act( (A .* A2)[256][K] @ W[N][K]^T + bias[N] )
// NT layout: both A and W are K-contiguous -> ideal for CDNA5 WMMA fragments.
// Tile: 256(M) x 64(N) x 32(K). 8 waves; wave = 128(M) x 16(N) = 8 wmma tiles.
// fp32 -> bf16 conversion during LDS staging; fp32 accumulation.
// Double-buffered LDS (ping-pong), one barrier per K-step; A-frag loads are
// software-pipelined one tile ahead of the WMMA chain.
// A2: optional elementwise multiplier on A (fuses xl*xr into the wo GEMM).
// ACT: 0=none, 1=silu, 2=softplus
// ===========================================================================
#define GEMM_BN 64
#define GEMM_BK 32

template <int ACT>
__global__ __launch_bounds__(256)
void gemm_nt_bf16(const float* __restrict__ A, const float* A2, int lda,
                  const float* __restrict__ W,
                  const float* __restrict__ bias,
                  float* __restrict__ C, int ldc,
                  int N, int K)
{
    __shared__ __bf16 As[2][256][GEMM_BK];      // 2 x 16 KB
    __shared__ __bf16 Bs[2][GEMM_BN][GEMM_BK];  // 2 x 4 KB

    const int tid   = threadIdx.x;
    const int nBase = blockIdx.x * GEMM_BN;
    const int wave  = tid >> 5;
    const int lane  = tid & 31;
    const int wM    = wave >> 2;       // 0..1 -> m offset wM*128
    const int wN    = wave & 3;        // 0..3 -> n offset wN*16
    const int lm    = lane & 15;
    const int l16   = (lane & 16) ? 1 : 0;

    const int bn = tid >> 2;           // B stage: row within tile
    const int bk = (tid & 3) * 8;      // B stage: col chunk
    const int gn = nBase + bn;
    const bool bvalid = (gn < N);

    v8f acc[8];
    const v8f vzero = {0.f, 0.f, 0.f, 0.f, 0.f, 0.f, 0.f, 0.f};
#pragma unroll
    for (int t = 0; t < 8; ++t) acc[t] = vzero;

    float4 ra[8];        // staged A row (32 fp32)
    float4 rb0, rb1;     // staged W chunk (8 fp32)

    // ---- global load of one K-tile into registers (with optional A2 mul)
    auto gload = [&](int k0) {
        const float* srcA = A + (size_t)tid * lda + k0;
#pragma unroll
        for (int j = 0; j < 8; ++j)
            ra[j] = *reinterpret_cast<const float4*>(srcA + 4 * j);
        if (A2) {
            const float* src2 = A2 + (size_t)tid * lda + k0;
#pragma unroll
            for (int j = 0; j < 8; ++j) {
                float4 m = *reinterpret_cast<const float4*>(src2 + 4 * j);
                ra[j].x *= m.x; ra[j].y *= m.y; ra[j].z *= m.z; ra[j].w *= m.w;
            }
        }
        if (bvalid) {
            const float* srcB = W + (size_t)gn * K + k0 + bk;
            rb0 = *reinterpret_cast<const float4*>(srcB);
            rb1 = *reinterpret_cast<const float4*>(srcB + 4);
        }
    };
    // ---- convert staged registers to bf16 and store into LDS buffer `b`
    auto sstore = [&](int b) {
#pragma unroll
        for (int j = 0; j < 8; ++j) {
            As[b][tid][4 * j + 0] = (__bf16)ra[j].x;
            As[b][tid][4 * j + 1] = (__bf16)ra[j].y;
            As[b][tid][4 * j + 2] = (__bf16)ra[j].z;
            As[b][tid][4 * j + 3] = (__bf16)ra[j].w;
        }
        if (bvalid) {
            Bs[b][bn][bk + 0] = (__bf16)rb0.x; Bs[b][bn][bk + 1] = (__bf16)rb0.y;
            Bs[b][bn][bk + 2] = (__bf16)rb0.z; Bs[b][bn][bk + 3] = (__bf16)rb0.w;
            Bs[b][bn][bk + 4] = (__bf16)rb1.x; Bs[b][bn][bk + 5] = (__bf16)rb1.y;
            Bs[b][bn][bk + 6] = (__bf16)rb1.z; Bs[b][bn][bk + 7] = (__bf16)rb1.w;
        } else {
#pragma unroll
            for (int j = 0; j < 8; ++j) Bs[b][bn][bk + j] = (__bf16)0.f;
        }
    };

    // ---- prologue: stage tile 0
    gload(0);
    sstore(0);
    __syncthreads();

    int buf = 0;
    for (int k0 = 0; k0 < K; k0 += GEMM_BK) {
        const bool more = (k0 + GEMM_BK) < K;
        if (more) {
            gload(k0 + GEMM_BK);                 // in flight under the WMMAs
            if (tid < GEMM_BN && bvalid && (k0 + 2 * GEMM_BK) < K)
                __builtin_prefetch(W + (size_t)(nBase + tid) * K + k0 + 2 * GEMM_BK, 0, 0);
        }

        // ---- B fragment: lane holds column n = wN*16+lm, K = l16*8+0..7 and 16+l16*8+0..7
        v8bf b0 = *reinterpret_cast<const v8bf*>(&Bs[buf][wN * 16 + lm][l16 * 8]);
        v8bf b1 = *reinterpret_cast<const v8bf*>(&Bs[buf][wN * 16 + lm][16 + l16 * 8]);
        v16bf bfrag = __builtin_shufflevector(b0, b1,
            0, 1, 2, 3, 4, 5, 6, 7, 8, 9, 10, 11, 12, 13, 14, 15);

        // ---- software-pipelined A-frag loads + 8 back-to-back WMMAs
        v8bf a0 = *reinterpret_cast<const v8bf*>(&As[buf][wM * 128 + lm][l16 * 8]);
        v8bf a1 = *reinterpret_cast<const v8bf*>(&As[buf][wM * 128 + lm][16 + l16 * 8]);
#pragma unroll
        for (int t = 0; t < 8; ++t) {
            v8bf n0 = a0, n1 = a1;
            if (t < 7) {
                const int m = wM * 128 + (t + 1) * 16 + lm;
                n0 = *reinterpret_cast<const v8bf*>(&As[buf][m][l16 * 8]);
                n1 = *reinterpret_cast<const v8bf*>(&As[buf][m][16 + l16 * 8]);
            }
            v16bf afrag = __builtin_shufflevector(a0, a1,
                0, 1, 2, 3, 4, 5, 6, 7, 8, 9, 10, 11, 12, 13, 14, 15);
            acc[t] = __builtin_amdgcn_wmma_f32_16x16x32_bf16(
                false, afrag, false, bfrag, (short)0, acc[t], false, false);
            a0 = n0; a1 = n1;
        }

        if (more) sstore(buf ^ 1);               // write other buffer, no conflict
        __syncthreads();
        buf ^= 1;
    }

    // ---- epilogue: bias + activation + store
    const int nIdx = nBase + wN * 16 + lm;
    if (nIdx < N) {
        const float bsv = bias[nIdx];
#pragma unroll
        for (int t = 0; t < 8; ++t) {
#pragma unroll
            for (int j = 0; j < 8; ++j) {
                const int m = wM * 128 + t * 16 + j + l16 * 8;
                float v = acc[t][j] + bsv;
                if (ACT == 1) v = v / (1.f + __expf(-v));                 // silu
                if (ACT == 2) v = (v > 20.f) ? v : log1pf(__expf(v));     // softplus
                C[(size_t)m * ldc + nIdx] = v;
            }
        }
    }
}

// ===========================================================================
// 3x3x3 conv, stride 1, pad 1, CH=64 -> 64. LDS-tiled: 4^3 output voxels.
// Input tile (+1 halo) staged in LDS as [ci][z][y][x] so that a wave
// (32 consecutive co values) broadcast-reads one LDS word per FMA.
//   in_eff = x (+ addx);  y = conv(in_eff)+b;  if leaky: y += leaky_relu(y);
//   if res: y += res
// ===========================================================================
__global__ __launch_bounds__(256)
void conv3x3_fused(const float* __restrict__ x, const float* addx,
                   const float* res,
                   const float* __restrict__ w, const float* __restrict__ b,
                   float* __restrict__ out, int D, int leaky)
{
    __shared__ float I[CH * 216];  // 64 * 6^3 floats = 55296 B

    const int tile  = blockIdx.x;
    const int batch = blockIdx.y;
    const int tpd   = D >> 2;
    const int tz = tile / (tpd * tpd);
    const int rem = tile % (tpd * tpd);
    const int ty = rem / tpd, tx = rem % tpd;
    const int z0 = tz * 4, y0 = ty * 4, x0 = tx * 4;
    const int tid = threadIdx.x;
    const int D2 = D * D, D3 = D2 * D;

    for (int idx = tid; idx < CH * 216; idx += 256) {
        const int ci = idx / 216;
        const int r  = idx % 216;
        const int z = r / 36, y = (r / 6) % 6, xx = r % 6;
        const int gz = z0 + z - 1, gy = y0 + y - 1, gx = x0 + xx - 1;
        float v = 0.f;
        if ((unsigned)gz < (unsigned)D && (unsigned)gy < (unsigned)D &&
            (unsigned)gx < (unsigned)D) {
            const size_t gi = (size_t)(batch * CH + ci) * D3 + gz * D2 + gy * D + gx;
            v = x[gi];
            if (addx) v += addx[gi];
        }
        I[idx] = v;
    }
    __syncthreads();

    const int co = tid & 63;
    const int g  = tid >> 6;  // voxel group 0..3 (16 voxels each)
    float acc[16];
#pragma unroll
    for (int vi = 0; vi < 16; ++vi) acc[vi] = 0.f;

    const float* wco = w + (size_t)co * (CH * 27);
    for (int ci = 0; ci < CH; ++ci) {
        const float* wc = wco + ci * 27;
        const float* Ic = &I[ci * 216];
#pragma unroll
        for (int t = 0; t < 27; ++t) {
            const int dz = t / 9, dy = (t / 3) % 3, dx = t % 3;
            const float wv = wc[t];
#pragma unroll
            for (int vi = 0; vi < 16; ++vi) {
                const int vl = g * 16 + vi;
                const int vz = vl >> 4, vy = (vl >> 2) & 3, vx = vl & 3;
                acc[vi] += wv * Ic[(vz + dz) * 36 + (vy + dy) * 6 + (vx + dx)];
            }
        }
    }

    const float bc = b[co];
#pragma unroll
    for (int vi = 0; vi < 16; ++vi) {
        const int vl = g * 16 + vi;
        const int vz = vl >> 4, vy = (vl >> 2) & 3, vx = vl & 3;
        float v = acc[vi] + bc;
        if (leaky) v = v + (v > 0.f ? v : 0.01f * v);   // y + leaky_relu(y)
        const size_t gi = (size_t)(batch * CH + co) * D3 +
                          (size_t)(z0 + vz) * D2 + (y0 + vy) * D + (x0 + vx);
        if (res) v += res[gi];
        out[gi] = v;
    }
}

// ===========================================================================
// 3x3x3 conv, stride 2, pad 1  (down-sampling conv)
// ===========================================================================
__global__ __launch_bounds__(256)
void conv3x3_s2(const float* __restrict__ x, const float* __restrict__ w,
                const float* __restrict__ b, float* __restrict__ out, int D)
{
    const int Do = D >> 1, Do2 = Do * Do, Do3 = Do2 * Do;
    const int total = BATCH * CH * Do3;
    const int i = blockIdx.x * 256 + threadIdx.x;
    if (i >= total) return;
    const int xo = i % Do, yo = (i / Do) % Do, zo = (i / Do2) % Do;
    const int co = (i / Do3) % CH, batch = i / (Do3 * CH);
    const int D2 = D * D, D3 = D2 * D;

    float acc = b[co];
    const float* wp = w + (size_t)co * (CH * 27);
    for (int ci = 0; ci < CH; ++ci) {
        const float* xp  = x + (size_t)(batch * CH + ci) * D3;
        const float* wp2 = wp + ci * 27;
#pragma unroll
        for (int dz = 0; dz < 3; ++dz) {
            const int iz = 2 * zo + dz - 1;
            if ((unsigned)iz >= (unsigned)D) continue;
#pragma unroll
            for (int dy = 0; dy < 3; ++dy) {
                const int iy = 2 * yo + dy - 1;
                if ((unsigned)iy >= (unsigned)D) continue;
#pragma unroll
                for (int dx = 0; dx < 3; ++dx) {
                    const int ix = 2 * xo + dx - 1;
                    if ((unsigned)ix >= (unsigned)D) continue;
                    acc += wp2[dz * 9 + dy * 3 + dx] * xp[iz * D2 + iy * D + ix];
                }
            }
        }
    }
    out[i] = acc;
}

// ===========================================================================
// 2x2x2 transpose conv (lhs_dilation=2, pad 1): each output voxel uses
// exactly one tap:  k = q&1 per axis, p = q>>1,  wt[co,ci,k] = w[ci,co,1-k]
// ===========================================================================
__global__ __launch_bounds__(256)
void convT2(const float* __restrict__ x, const float* __restrict__ w,
            const float* __restrict__ b, float* __restrict__ out, int Din)
{
    const int Do = Din * 2, Do2 = Do * Do, Do3 = Do2 * Do;
    const int total = BATCH * CH * Do3;
    const int i = blockIdx.x * 256 + threadIdx.x;
    if (i >= total) return;
    const int qx = i % Do, qy = (i / Do) % Do, qz = (i / Do2) % Do;
    const int co = (i / Do3) % CH, batch = i / (Do3 * CH);
    const int ko = (qz & 1) * 4 + (qy & 1) * 2 + (qx & 1);
    const int px = qx >> 1, py = qy >> 1, pz = qz >> 1;
    const int Di2 = Din * Din, Di3 = Di2 * Din;

    float acc = b[co];
    const size_t xoff = (size_t)batch * CH * Di3 + (size_t)pz * Di2 + py * Din + px;
    for (int ci = 0; ci < CH; ++ci)
        acc += w[(size_t)(ci * CH + co) * 8 + ko] * x[xoff + (size_t)ci * Di3];
    out[i] = acc;
}

// ===========================================================================
// LayerNorm over spatial dim L per (b,ch) row; per-position gains g[l], b[l]
// ===========================================================================
__global__ __launch_bounds__(256)
void layernorm_k(const float* __restrict__ x, const float* __restrict__ g,
                 const float* __restrict__ bt, float* __restrict__ out, int L)
{
    __shared__ float s1[256], s2[256];
    const int row = blockIdx.x;  // 0..255 = b*CH+ch
    const float* xr = x + (size_t)row * L;
    float sum = 0.f, sq = 0.f;
    for (int i = threadIdx.x; i < L; i += 256) {
        const float v = xr[i];
        sum += v; sq += v * v;
    }
    s1[threadIdx.x] = sum; s2[threadIdx.x] = sq;
    __syncthreads();
    for (int o = 128; o > 0; o >>= 1) {
        if (threadIdx.x < o) {
            s1[threadIdx.x] += s1[threadIdx.x + o];
            s2[threadIdx.x] += s2[threadIdx.x + o];
        }
        __syncthreads();
    }
    const float mu   = s1[0] / (float)L;
    const float var  = s2[0] / (float)L - mu * mu;
    const float rstd = rsqrtf(var + 1e-5f);
    for (int i = threadIdx.x; i < L; i += 256)
        out[(size_t)row * L + i] = (xr[i] - mu) * rstd * g[i] + bt[i];
}

// ===========================================================================
// conv1d over L (kernel 3, pad 1, CH->CH mix) fused with silu
// grid: (ceil(L/256), CH, BATCH)
// ===========================================================================
__global__ __launch_bounds__(256)
void conv1d_silu(const float* __restrict__ x, const float* __restrict__ w,
                 const float* __restrict__ b, float* __restrict__ out, int L)
{
    __shared__ float ws[CH * 3];
    const int co = blockIdx.y, batch = blockIdx.z;
    if (threadIdx.x < CH * 3) ws[threadIdx.x] = w[co * CH * 3 + threadIdx.x];
    __syncthreads();
    const int l = blockIdx.x * 256 + threadIdx.x;
    if (l >= L) return;
    float acc = b[co];
    const float* xb = x + (size_t)batch * CH * L;
    for (int ci = 0; ci < CH; ++ci) {
        const float* xc = xb + (size_t)ci * L;
        const float x0 = (l > 0)     ? xc[l - 1] : 0.f;
        const float x1 = xc[l];
        const float x2 = (l < L - 1) ? xc[l + 1] : 0.f;
        acc += ws[ci * 3 + 0] * x0 + ws[ci * 3 + 1] * x1 + ws[ci * 3 + 2] * x2;
    }
    acc = acc / (1.f + __expf(-acc));  // silu
    out[((size_t)batch * CH + co) * L + l] = acc;
}

// ===========================================================================
// Selective-scan over the channel axis (seq len = 64), state NSTATE=16.
// A[n] = -(n+1)  =>  dA = exp(-(n+1)*delta) = pow(exp(-delta), n+1).
// Bm/Cm staged in LDS per batch.  out = scan_y + xl * D[l]
// grid: (ceil(L/256), BATCH)
// ===========================================================================
__global__ __launch_bounds__(256)
void ssm_scan(const float* __restrict__ xl, const float* __restrict__ delta,
              const float* __restrict__ xd,  // row stride 96: Bm @+64, Cm @+80
              const float* __restrict__ Dv, float* __restrict__ out, int L)
{
    __shared__ float Bs[CH][16], Cs[CH][16];
    const int batch = blockIdx.y;
    for (int i = threadIdx.x; i < CH * 16; i += 256) {
        const int c = i >> 4, n = i & 15;
        const size_t rb = ((size_t)batch * CH + c) * 96;
        Bs[c][n] = xd[rb + 64 + n];
        Cs[c][n] = xd[rb + 80 + n];
    }
    __syncthreads();
    const int l = blockIdx.x * 256 + threadIdx.x;
    if (l >= L) return;

    float h[16];
#pragma unroll
    for (int n = 0; n < 16; ++n) h[n] = 0.f;
    const float dv = Dv[l];

    for (int c = 0; c < CH; ++c) {
        const size_t off = ((size_t)batch * CH + c) * L + l;
        const float dt  = delta[off];
        const float xv  = xl[off];
        const float dbx = dt * xv;
        const float e1  = __expf(-dt);
        float p = 1.f, y = 0.f;
#pragma unroll
        for (int n = 0; n < 16; ++n) {
            p *= e1;                            // exp(-(n+1)*dt)
            h[n] = p * h[n] + dbx * Bs[c][n];
            y += h[n] * Cs[c][n];
        }
        out[off] = y + xv * dv;
    }
}

// ===========================================================================
// Elementwise add (residual / skip connections)
// ===========================================================================
__global__ void ew_add(const float* a, const float* b, float* c, int n)
{
    const int i = blockIdx.x * 256 + threadIdx.x;
    if (i < n) c[i] = a[i] + b[i];
}

// ===========================================================================
// Host-side orchestration
// ===========================================================================
struct LP {
    const float *conv_w, *conv_b, *c1w, *c1b, *ln_g, *ln_b;
    const float *wl, *bl, *wr, *br, *wo, *bo, *wx, *bx, *wdt, *bdt, *Dv;
};

static void readLayer(void* const* d, int& i, LP& p)
{
    p.conv_w = (const float*)d[i++]; p.conv_b = (const float*)d[i++];
    p.c1w    = (const float*)d[i++]; p.c1b    = (const float*)d[i++];
    p.ln_g   = (const float*)d[i++]; p.ln_b   = (const float*)d[i++];
    p.wl     = (const float*)d[i++]; p.bl     = (const float*)d[i++];
    p.wr     = (const float*)d[i++]; p.br     = (const float*)d[i++];
    p.wo     = (const float*)d[i++]; p.bo     = (const float*)d[i++];
    p.wx     = (const float*)d[i++]; p.bx     = (const float*)d[i++];
    p.wdt    = (const float*)d[i++]; p.bdt    = (const float*)d[i++];
    p.Dv     = (const float*)d[i++];
}

static void run_umamba(const float* xin, const LP& p, int D,
                       float* S0, float* S1, float* S2, float* S3, float* XD,
                       float* outb, hipStream_t s)
{
    const int L   = D * D * D;
    const int tpd = D >> 2;
    const dim3 gridC(tpd * tpd * tpd, BATCH);
    const int gN = (L + GEMM_BN - 1) / GEMM_BN;

    float* XC = S0;
    float* X2 = S1;
    // xc = conv(x)+b; xc += leaky(xc)
    conv3x3_fused<<<gridC, 256, 0, s>>>(xin, nullptr, nullptr, p.conv_w, p.conv_b, XC, D, 1);
    // x2 = conv(xc + x)+b; x2 += leaky(x2); x2 += xc
    conv3x3_fused<<<gridC, 256, 0, s>>>(XC, xin, XC, p.conv_w, p.conv_b, X2, D, 1);

    float* YLN = S2;
    layernorm_k<<<256, 256, 0, s>>>(X2, p.ln_g, p.ln_b, YLN, L);

    float* XL = S1;  // X2 dead
    gemm_nt_bf16<0><<<gN, 256, 0, s>>>(YLN, nullptr, L, p.wl, p.bl, XL, L, L, L);

    float* XL2 = S3;
    dim3 g1((L + 255) / 256, CH, BATCH);
    conv1d_silu<<<g1, 256, 0, s>>>(XL, p.c1w, p.c1b, XL2, L);

    // xd = xl2 @ wx.T + bx  (256 x 96)
    gemm_nt_bf16<0><<<(96 + GEMM_BN - 1) / GEMM_BN, 256, 0, s>>>(
        XL2, nullptr, L, p.wx, p.bx, XD, 96, 96, L);

    float* DELTA = S0;  // XC dead
    // delta = softplus(xd[:, :64] @ wdt.T + bdt)
    gemm_nt_bf16<2><<<gN, 256, 0, s>>>(XD, nullptr, 96, p.wdt, p.bdt, DELTA, L, L, 64);

    float* YS = S1;  // XL dead
    dim3 gs((L + 255) / 256, BATCH);
    ssm_scan<<<gs, 256, 0, s>>>(XL2, DELTA, XD, p.Dv, YS, L);

    float* XR = S0;  // DELTA dead
    gemm_nt_bf16<1><<<gN, 256, 0, s>>>(YLN, nullptr, L, p.wr, p.br, XR, L, L, L);

    // out = (ys .* xr) @ wo.T + bo   (elementwise product fused into staging)
    gemm_nt_bf16<0><<<gN, 256, 0, s>>>(YS, XR, L, p.wo, p.bo, outb, L, L, L);
}

extern "C" void kernel_launch(void* const* d_in, const int* in_sizes, int n_in,
                              void* d_out, int out_size, void* d_ws, size_t ws_size,
                              hipStream_t stream)
{
    (void)in_sizes; (void)n_in; (void)out_size; (void)ws_size;

    const float* x = (const float*)d_in[0];
    int i = 1;
    LP down0, down1, up0, up1;
    readLayer(d_in, i, down0);
    const float* dw0 = (const float*)d_in[i++]; const float* db0 = (const float*)d_in[i++];
    readLayer(d_in, i, down1);
    const float* dw1 = (const float*)d_in[i++]; const float* db1 = (const float*)d_in[i++];
    readLayer(d_in, i, up0);
    const float* uw0 = (const float*)d_in[i++]; const float* ub0 = (const float*)d_in[i++];
    readLayer(d_in, i, up1);
    const float* uw1 = (const float*)d_in[i++]; const float* ub1 = (const float*)d_in[i++];

    // ---- workspace arena ----
    char* wsb = (char*)d_ws;
    size_t off = 0;
    auto alloc = [&](size_t bytes) -> float* {
        float* p = (float*)(wsb + off);
        off += (bytes + 255) & ~(size_t)255;
        return p;
    };
    const size_t ACT16 = (size_t)BATCH * CH * 4096 * sizeof(float);  // 4 MB
    const size_t ACT8  = (size_t)BATCH * CH * 512  * sizeof(float);  // 512 KB
    const size_t ACT4  = (size_t)BATCH * CH * 64   * sizeof(float);  // 64 KB

    float* S0 = alloc(ACT16);
    float* S1 = alloc(ACT16);
    float* S2 = alloc(ACT16);
    float* S3 = alloc(ACT16);
    float* XD = alloc((size_t)256 * 96 * sizeof(float));
    float* G1 = alloc(ACT16);
    float* H0 = alloc(ACT8);
    float* G2 = alloc(ACT8);
    float* H1 = alloc(ACT4);
    float* T0 = alloc(ACT4);
    float* G3 = alloc(ACT4);
    float* T1 = alloc(ACT8);
    float* T2 = alloc(ACT8);
    float* G4 = alloc(ACT8);

    // ---- down 0 : 16^3 (L=4096) ----
    run_umamba(x, down0, 16, S0, S1, S2, S3, XD, G1, stream);
    {
        const int tot = BATCH * CH * 8 * 8 * 8;
        conv3x3_s2<<<(tot + 255) / 256, 256, 0, stream>>>(G1, dw0, db0, H0, 16);
    }
    // ---- down 1 : 8^3 (L=512) ----
    run_umamba(H0, down1, 8, S0, S1, S2, S3, XD, G2, stream);
    {
        const int tot = BATCH * CH * 4 * 4 * 4;
        conv3x3_s2<<<(tot + 255) / 256, 256, 0, stream>>>(G2, dw1, db1, H1, 8);
    }
    // ---- up 0 : x = x + hist1 (== 2*H1), umamba at 4^3 (L=64), upsample ----
    {
        const int n = BATCH * CH * 64;
        ew_add<<<(n + 255) / 256, 256, 0, stream>>>(H1, H1, T0, n);
    }
    run_umamba(T0, up0, 4, S0, S1, S2, S3, XD, G3, stream);
    {
        const int tot = BATCH * CH * 8 * 8 * 8;
        convT2<<<(tot + 255) / 256, 256, 0, stream>>>(G3, uw0, ub0, T1, 4);
    }
    // ---- up 1 : x = x + hist0, umamba at 8^3 (L=512), upsample -> out ----
    {
        const int n = BATCH * CH * 512;
        ew_add<<<(n + 255) / 256, 256, 0, stream>>>(T1, H0, T2, n);
    }
    run_umamba(T2, up1, 8, S0, S1, S2, S3, XD, G4, stream);
    {
        const int tot = BATCH * CH * 16 * 16 * 16;
        convT2<<<(tot + 255) / 256, 256, 0, stream>>>(G4, uw1, ub1, (float*)d_out, 8);
    }
}
